// FlowmatchingActionHead_45174466019822
// MI455X (gfx1250) — matrix operand order repaired and checked by reading, not compiled, not run on hardware
//
#include <hip/hip_runtime.h>
#include <math.h>

// ---------------------------------------------------------------------------
// FlowmatchingActionHead for MI455X (gfx1250), wave32 + WMMA bf16 path.
//
// Heavy GEMMs (Mamba in/out projections, final out_proj) run through a
// 128x128x32-tiled WMMA kernel (v_wmma_f32_16x16x32_bf16, f32 accumulate)
// with double-buffered LDS staging (global loads for tile k+1 issued before
// the WMMAs of tile k -> LOADcnt latency hides under compute) and a
// branch-free fast path for full tiles (remainder rows get a separate
// guarded launch). Workload is HBM-bound (~10 GB activation traffic vs
// 2.5 TFLOP -> ~0.45 ms floor at 23.3 TB/s), so epilogues (bias/residual)
// are fused and everything else is single-pass.
// ---------------------------------------------------------------------------

typedef __attribute__((ext_vector_type(16))) __bf16        bf16x16;
typedef __attribute__((ext_vector_type(8)))  float         f32x8;
typedef __attribute__((ext_vector_type(8)))  unsigned int  u32x8;

#define BM 128
#define BN 128
#define BK 32
#define LDS_STRIDE 34   // BK + 2 pad (68B rows keep u32 alignment, spread banks)

__device__ __forceinline__ unsigned short f2bf(float f) {
  unsigned int u = __builtin_bit_cast(unsigned int, f);
  u += 0x7FFFu + ((u >> 16) & 1u);      // round-to-nearest-even
  return (unsigned short)(u >> 16);
}
__device__ __forceinline__ unsigned int f2bf2(float lo, float hi) {
  return (unsigned int)f2bf(lo) | ((unsigned int)f2bf(hi) << 16);
}

// C[M,N] = A[M,K] @ B[K,N] (+ bias[N]) (+ C if addC)   -- K%32==0, N%128==0
// GUARD=false: all 128 rows of this block are valid (fast path, no branches).
template <bool GUARD>
__global__ __launch_bounds__(256) void gemm_wmma_bf16(
    const float* __restrict__ A, const float* __restrict__ Bw,
    float* __restrict__ C, const float* __restrict__ bias,
    int M, int N, int K, int addC, int row_off)
{
  __shared__ unsigned short lA[2][BM * LDS_STRIDE];
  __shared__ unsigned short lB[2][BN * LDS_STRIDE];   // transposed: [n][k]

  const int tid   = threadIdx.x;
  const int lane  = tid & 31;
  const int wave  = tid >> 5;
  const int wrow  = wave >> 1;        // 0..3 -> 32 rows per wave
  const int wcol  = wave & 1;         // 0..1 -> 64 cols per wave
  const int row0  = row_off + blockIdx.y * BM;
  const int col0  = blockIdx.x * BN;
  const int mloc  = lane & 15;
  const int khalf = lane >> 4;

  f32x8 acc[2][4] = {};
  float4 ra[4], rb[4];

  // issue global loads for the BK-slice starting at kk into registers
  auto loadG = [&](int kk) {
#pragma unroll
    for (int it = 0; it < 4; ++it) {
      int idx = tid + it * 256;            // 1024 float4 slots
      int r   = idx >> 3;                  // A: 8 float4 per row
      int kv  = (idx & 7) << 2;
      if (!GUARD || (row0 + r) < M)
        ra[it] = *(const float4*)(A + (size_t)(row0 + r) * K + kk + kv);
      else
        ra[it] = make_float4(0.f, 0.f, 0.f, 0.f);
      int kr = idx >> 5;                   // B: 32 float4 per k-row
      int nv = (idx & 31) << 2;
      rb[it] = *(const float4*)(Bw + (size_t)(kk + kr) * N + col0 + nv);
    }
  };
  // convert + store staged registers into LDS buffer p
  auto storeL = [&](int p) {
#pragma unroll
    for (int it = 0; it < 4; ++it) {
      int idx = tid + it * 256;
      int r   = idx >> 3;
      int kv  = (idx & 7) << 2;
      unsigned int* d = (unsigned int*)&lA[p][r * LDS_STRIDE + kv];
      d[0] = f2bf2(ra[it].x, ra[it].y);
      d[1] = f2bf2(ra[it].z, ra[it].w);
      int kr = idx >> 5;
      int nv = (idx & 31) << 2;
      lB[p][(nv + 0) * LDS_STRIDE + kr] = f2bf(rb[it].x);
      lB[p][(nv + 1) * LDS_STRIDE + kr] = f2bf(rb[it].y);
      lB[p][(nv + 2) * LDS_STRIDE + kr] = f2bf(rb[it].z);
      lB[p][(nv + 3) * LDS_STRIDE + kr] = f2bf(rb[it].w);
    }
  };

  const int nk = K / BK;
  loadG(0);
  storeL(0);
  __syncthreads();

  for (int t = 0; t < nk; ++t) {
    const int p = t & 1;
    if (t + 1 < nk) loadG((t + 1) * BK);   // overlap with WMMAs below

    // ---- fragments per ISA 7.12.2 VGPR layouts ----
    bf16x16 afrag[2], bfrag[4];
#pragma unroll
    for (int i = 0; i < 2; ++i) {
      int m = wrow * 32 + i * 16 + mloc;
      u32x8 u;
#pragma unroll
      for (int v = 0; v < 8; ++v) {
        int kb = ((v < 4) ? (2 * v) : (16 + 2 * (v - 4))) + khalf * 8;
        u[v] = *(const unsigned int*)&lA[p][m * LDS_STRIDE + kb];
      }
      afrag[i] = __builtin_bit_cast(bf16x16, u);
    }
#pragma unroll
    for (int j = 0; j < 4; ++j) {
      int n = wcol * 64 + j * 16 + mloc;
      u32x8 u;
#pragma unroll
      for (int v = 0; v < 8; ++v) {
        int kb = ((v < 4) ? (2 * v) : (16 + 2 * (v - 4))) + khalf * 8;
        u[v] = *(const unsigned int*)&lB[p][n * LDS_STRIDE + kb];
      }
      bfrag[j] = __builtin_bit_cast(bf16x16, u);
    }
#pragma unroll
    for (int i = 0; i < 2; ++i)
#pragma unroll
      for (int j = 0; j < 4; ++j)
        acc[i][j] = __builtin_amdgcn_wmma_f32_16x16x32_bf16(
            false, afrag[i], false, bfrag[j], (short)0, acc[i][j], false, false);

    if (t + 1 < nk) storeL(1 - p);         // fill the other buffer
    __syncthreads();
  }

  // ---- epilogue: D layout = lane(N), vgpr r -> M = khalf*8 + r ----
#pragma unroll
  for (int i = 0; i < 2; ++i) {
#pragma unroll
    for (int j = 0; j < 4; ++j) {
      int col   = col0 + wcol * 64 + j * 16 + mloc;
      int rbase = row0 + wrow * 32 + i * 16 + khalf * 8;
      float bb = bias ? bias[col] : 0.f;
#pragma unroll
      for (int r = 0; r < 8; ++r) {
        int row = rbase + r;
        if (!GUARD || row < M) {
          size_t o = (size_t)row * N + col;
          float v = acc[i][j][r] + bb;
          if (addC) v += C[o];
          C[o] = v;
        }
      }
    }
  }
}

// y = silu(depthwise_causal_conv(xc)) * sigmoid(z);  xz = [xc | z] row-major
__global__ __launch_bounds__(256) void conv_gate(
    const float* __restrict__ xz, const float* __restrict__ w,
    const float* __restrict__ wb, float* __restrict__ y,
    int T, int Di, long long total)
{
  long long e = (long long)blockIdx.x * 256 + threadIdx.x;
  if (e >= total) return;
  int c = (int)(e % Di);
  long long row = e / Di;
  int t = (int)(row % T);
  const long long stride = 2LL * Di;
  float acc = wb[c];
#pragma unroll
  for (int k = 0; k < 4; ++k) {
    int dt = k - 3;
    if (t + dt >= 0)
      acc += xz[(row + dt) * stride + c] * w[k * Di + c];
  }
  float z = xz[row * stride + Di + c];
  float sa = 1.f / (1.f + __expf(-acc));
  float sz = 1.f / (1.f + __expf(-z));
  y[e] = acc * sa * sz;
}

__global__ __launch_bounds__(256) void layernorm(
    const float* __restrict__ x, const float* __restrict__ g,
    const float* __restrict__ b, float* __restrict__ y, int D)
{
  __shared__ float red[256];
  int row = blockIdx.x, tid = threadIdx.x;
  const float* xr = x + (size_t)row * D;
  float* yr = y + (size_t)row * D;
  float s = 0.f;
  for (int i = tid; i < D; i += 256) s += xr[i];
  red[tid] = s; __syncthreads();
  for (int o = 128; o > 0; o >>= 1) { if (tid < o) red[tid] += red[tid + o]; __syncthreads(); }
  float mu = red[0] / D;
  __syncthreads();
  float v = 0.f;
  for (int i = tid; i < D; i += 256) { float d = xr[i] - mu; v += d * d; }
  red[tid] = v; __syncthreads();
  for (int o = 128; o > 0; o >>= 1) { if (tid < o) red[tid] += red[tid + o]; __syncthreads(); }
  float inv = rsqrtf(red[0] / D + 1e-5f);
  for (int i = tid; i < D; i += 256) yr[i] = (xr[i] - mu) * inv * g[i] + b[i];
}

// category-specific linear: out[s,n] = act(x[s,:] @ W[id(s)] + b[id(s)])
__global__ __launch_bounds__(256) void catlin(
    const float* __restrict__ x, const float* __restrict__ W,
    const float* __restrict__ b, const int* __restrict__ ids,
    float* __restrict__ out, int S, int tpb, int K, int N, int act)
{
  int e = blockIdx.x * 256 + threadIdx.x;
  if (e >= S * N) return;
  int s = e / N, n = e % N;
  int id = ids[s / tpb];
  const float* xr = x + (size_t)s * K;
  const float* wc = W + (size_t)id * K * N + n;
  float acc = b[(size_t)id * N + n];
  for (int k = 0; k < K; ++k) acc += xr[k] * wc[(size_t)k * N];
  if (act == 1) acc = fmaxf(acc, 0.f);
  else if (act == 2) acc = acc / (1.f + __expf(-acc));
  out[e] = acc;
}

// xcat[s, 0:1536] = a[s], xcat[s, 1536:3072] = interleaved sin/cos pos enc
__global__ __launch_bounds__(256) void build_xcat(
    const float* __restrict__ a, const int* __restrict__ ts,
    float* __restrict__ xc)
{
  int e = blockIdx.x * 256 + threadIdx.x;
  if (e >= 256 * 3072) return;
  int s = e / 3072, c = e % 3072;
  float v;
  if (c < 1536) {
    v = a[s * 1536 + c];
  } else {
    int cc = c - 1536;
    int i = cc >> 1;
    float div = __expf(-logf(10000.f) * (float)i / 768.f);
    float ang = (float)ts[s >> 4] * div;
    v = (cc & 1) ? __cosf(ang) : __sinf(ang);
  }
  xc[e] = v;
}

// tokens = [vl(512) | future(32) | state(1)+pos | actions(16)+pos] per batch
__global__ __launch_bounds__(256) void assemble_tokens(
    const float* __restrict__ vl, const float* __restrict__ fut,
    const float* __restrict__ sf, const float* __restrict__ af,
    const float* __restrict__ pos, float* __restrict__ tok, long long total)
{
  long long e = (long long)blockIdx.x * 256 + threadIdx.x;
  if (e >= total) return;
  int c = (int)(e % 1536);
  long long rr = e / 1536;
  int r = (int)(rr % 561);
  int bi = (int)(rr / 561);
  float v;
  if (r < 512)       v = vl[((long long)bi * 512 + r) * 1536 + c];
  else if (r < 544)  v = fut[(r - 512) * 1536 + c];
  else if (r == 544) v = sf[bi * 1536 + c] + pos[c];
  else               v = af[((long long)bi * 16 + (r - 545)) * 1536 + c]
                          + pos[(r - 544) * 1536 + c];
  tok[e] = v;
}

// gather last 16 tokens of each batch from the projected output [B,561,1024]
__global__ __launch_bounds__(256) void extract_dec(
    const float* __restrict__ outp, float* __restrict__ din)
{
  int e = blockIdx.x * 256 + threadIdx.x;
  if (e >= 256 * 1024) return;
  int c = e % 1024, s = e / 1024;
  int b = s / 16, t = s % 16;
  din[e] = outp[((size_t)b * 561 + 545 + t) * 1024 + c];
}

// ---------------------------------------------------------------------------
// Param leaf layout (jax pytree: dict keys sorted; lists in order):
//  d_in[0..4]: backbone, state, actions, timesteps, embodiment_id
//  d_in[5+i]:
//   0 act_W1  1 act_W2  2 act_W3  3 act_b1  4 act_b2  5 act_b3
//   6 dec_W1  7 dec_W2  8 dec_b1  9 dec_b2  10 future_tokens
//   11..34 model_layers[l]{conv_b,conv_w,in_proj,out_proj}
//   35 model_nb 36 model_ng 37 out_proj_b 38 out_proj_w 39 pos_embed
//   40 state_W1 41 state_W2 42 state_b1 43 state_b2
//   44..59 vl_layers[l]{conv_b,conv_w,in_proj,out_proj}
//   60 vl_nb 61 vl_ng 62 vlln_b 63 vlln_g
// ---------------------------------------------------------------------------
extern "C" void kernel_launch(void* const* d_in, const int* in_sizes, int n_in,
                              void* d_out, int out_size, void* d_ws, size_t ws_size,
                              hipStream_t stream) {
  (void)in_sizes; (void)n_in; (void)out_size; (void)ws_size;
  const float* backbone = (const float*)d_in[0];
  const float* state    = (const float*)d_in[1];
  const float* actions  = (const float*)d_in[2];
  const int*   tsteps   = (const int*)d_in[3];
  const int*   embid    = (const int*)d_in[4];
  auto prm = [&](int i) { return (const float*)d_in[5 + i]; };

  const int Bn = 16, TVL = 512, D = 1536, DI = 3072, TTOT = 561;
  const int MVL = Bn * TVL;     // 8192
  const int MTK = Bn * TTOT;    // 8976

  char* ws = (char*)d_ws;
  size_t off = 0;
  auto alloc = [&](size_t bytes) {
    float* p = (float*)(ws + off);
    off += (bytes + 255) & ~(size_t)255;
    return p;
  };
  float* bufVL  = alloc((size_t)MVL * D * 4);       // VL activations
  float* bufTok = alloc((size_t)MTK * D * 4);       // model-token activations
  float* bufXZ  = alloc((size_t)MTK * 4 * D * 4);   // in_proj output
  float* bufY   = alloc((size_t)MTK * 2 * D * 4);   // gated conv output
  float* bufOut = alloc((size_t)MTK * 1024 * 4);    // final projection
  float* stateH = alloc((size_t)16 * 1024 * 4);
  float* sfBuf  = alloc((size_t)16 * 1536 * 4);
  float* aBuf   = alloc((size_t)256 * 1536 * 4);
  float* xcat   = alloc((size_t)256 * 3072 * 4);
  float* afBuf  = alloc((size_t)256 * 1536 * 4);
  float* decIn  = alloc((size_t)256 * 1024 * 4);
  float* decH   = alloc((size_t)256 * 1024 * 4);

  auto gemm = [&](const float* Ap, const float* Bp, float* Cp, const float* bias,
                  int M, int N, int K, int addC) {
    int mfull = M / BM;
    if (mfull > 0)
      gemm_wmma_bf16<false><<<dim3((unsigned)(N / BN), (unsigned)mfull), 256, 0, stream>>>(
          Ap, Bp, Cp, bias, M, N, K, addC, 0);
    if (M % BM)
      gemm_wmma_bf16<true><<<dim3((unsigned)(N / BN), 1), 256, 0, stream>>>(
          Ap, Bp, Cp, bias, M, N, K, addC, mfull * BM);
  };

  // ---- VL path: LN -> 4 Mamba blocks -> LN ----
  layernorm<<<MVL, 256, 0, stream>>>(backbone, prm(63), prm(62), bufVL, D);
  for (int l = 0; l < 4; ++l) {
    int base = 44 + 4 * l;
    gemm(bufVL, prm(base + 2), bufXZ, nullptr, MVL, 4 * D, D, 0);
    long long tot = (long long)MVL * DI;
    conv_gate<<<(unsigned)((tot + 255) / 256), 256, 0, stream>>>(
        bufXZ, prm(base + 1), prm(base + 0), bufY, TVL, DI, tot);
    gemm(bufY, prm(base + 3), bufVL, nullptr, MVL, D, DI, 1);  // + residual
  }
  layernorm<<<MVL, 256, 0, stream>>>(bufVL, prm(61), prm(60), bufVL, D);

  // ---- state encoder (CategorySpecificMLP) ----
  catlin<<<(16 * 1024 + 255) / 256, 256, 0, stream>>>(
      state, prm(40), prm(42), embid, stateH, 16, 1, 64, 1024, 1);
  catlin<<<(16 * 1536 + 255) / 256, 256, 0, stream>>>(
      stateH, prm(41), prm(43), embid, sfBuf, 16, 1, 1024, 1536, 0);

  // ---- action encoder ----
  catlin<<<(256 * 1536 + 255) / 256, 256, 0, stream>>>(
      actions, prm(0), prm(3), embid, aBuf, 256, 16, 32, 1536, 0);
  build_xcat<<<(256 * 3072 + 255) / 256, 256, 0, stream>>>(aBuf, tsteps, xcat);
  catlin<<<(256 * 1536 + 255) / 256, 256, 0, stream>>>(
      xcat, prm(1), prm(4), embid, aBuf, 256, 16, 3072, 1536, 2);   // silu
  catlin<<<(256 * 1536 + 255) / 256, 256, 0, stream>>>(
      aBuf, prm(2), prm(5), embid, afBuf, 256, 16, 1536, 1536, 0);

  // ---- assemble [vl | future | state+pos | actions+pos] ----
  long long ttot = (long long)MTK * D;
  assemble_tokens<<<(unsigned)((ttot + 255) / 256), 256, 0, stream>>>(
      bufVL, prm(10), sfBuf, afBuf, prm(39), bufTok, ttot);

  // ---- model: 6 Mamba blocks -> LN -> out_proj ----
  for (int l = 0; l < 6; ++l) {
    int base = 11 + 4 * l;
    gemm(bufTok, prm(base + 2), bufXZ, nullptr, MTK, 4 * D, D, 0);
    long long tot = (long long)MTK * DI;
    conv_gate<<<(unsigned)((tot + 255) / 256), 256, 0, stream>>>(
        bufXZ, prm(base + 1), prm(base + 0), bufY, TTOT, DI, tot);
    gemm(bufY, prm(base + 3), bufTok, nullptr, MTK, D, DI, 1);
  }
  layernorm<<<MTK, 256, 0, stream>>>(bufTok, prm(36), prm(35), bufTok, D);
  gemm(bufTok, prm(38), bufOut, prm(37), MTK, 1024, D, 0);

  // ---- decode last 16 tokens ----
  extract_dec<<<(256 * 1024 + 255) / 256, 256, 0, stream>>>(bufOut, decIn);
  catlin<<<(256 * 1024 + 255) / 256, 256, 0, stream>>>(
      decIn, prm(6), prm(8), embid, decH, 256, 16, 1024, 1024, 1);
  catlin<<<(256 * 32 + 255) / 256, 256, 0, stream>>>(
      decH, prm(7), prm(9), embid, (float*)d_out, 256, 16, 1024, 32, 0);
}
